// MultiHeadedSelfAttention_60567628808855
// MI455X (gfx1250) — compile-verified, matrix operand order checked by reading, not run on hardware
//
#include <hip/hip_runtime.h>

// ---------------------------------------------------------------------------
// MultiHeadedSelfAttention for MI455X (gfx1250, wave32, WMMA bf16)
// B=2, L=2048, D=1024, H=16, P=64
// ---------------------------------------------------------------------------

typedef __bf16 bf16_t;
typedef __attribute__((ext_vector_type(16))) __bf16 bf16x16;
typedef __attribute__((ext_vector_type(8)))  float  f32x8;

constexpr int Bc = 2, Lc = 2048, Dc = 1024, Hc = 16, Pc = 64;
constexpr size_t QKV_BYTES = (size_t)Bc * Hc * Lc * Pc * sizeof(bf16_t); // 8 MiB

__device__ __forceinline__ f32x8 wmma_bf16(bf16x16 a, bf16x16 b, f32x8 c) {
  // D = A(16x32 bf16) x B(32x16 bf16) + C(16x16 f32)
  return __builtin_amdgcn_wmma_f32_16x16x32_bf16(
      false, a, false, b, (short)0, c, false, false);
}

__device__ __forceinline__ unsigned pack2bf(float a, float b) {
  __bf16 x = (__bf16)a, y = (__bf16)b;
  unsigned short ux = __builtin_bit_cast(unsigned short, x);
  unsigned short uy = __builtin_bit_cast(unsigned short, y);
  return (unsigned)ux | ((unsigned)uy << 16);
}

// Async global->LDS 16-byte copy (GLOBAL_LOAD_ASYNC_TO_LDS_B128, ASYNCcnt).
// lds_off: per-lane LDS byte address (low 32 bits of generic shared pointer).
__device__ __forceinline__ void async_g2lds_b128(unsigned lds_off, const void* g) {
  asm volatile("global_load_async_to_lds_b128 %0, %1, off"
               :: "v"(lds_off), "v"(g) : "memory");
}
__device__ __forceinline__ void wait_async0() {
  asm volatile("s_wait_asynccnt 0x0" ::: "memory");
}

// Load a 16x32 bf16 fragment (A-frag layout; identical pattern serves the
// B operand when the B matrix is "rows over the contraction axis").
// Lane l (m = l&15, c = l>>4): two contiguous 16-byte chunks:
//   K = c*8 + {0..7}  and  K = c*8 + 16 + {0..7}   (bf16 pairs per dword)
__device__ __forceinline__ bf16x16 ld_frag(const bf16_t* base, int r0, int stride,
                                           int k0, int lane) {
  const int m = lane & 15, c = lane >> 4;
  const bf16_t* p = base + (size_t)(r0 + m) * stride + k0 + c * 8;
  union { bf16x16 v; uint4 q[2]; } u;
  u.q[0] = *(const uint4*)(p);
  u.q[1] = *(const uint4*)(p + 16);
  return u.v;
}

// ---------------------------------------------------------------------------
// Kernel 1: fused QKV projection.  Y = x @ W^T + bias
// grid = (M/128, N/128, 3) ; z selects {Q, K, V}
// WG tile 128x128, wave tile 64x32 (8 WMMA per K-step).
// Q,K written bf16 as [B,H,L,P]; V written transposed [B,H,P,L].
// (Staging stays in VALU: must convert fp32 -> bf16 en route.)
// ---------------------------------------------------------------------------
__global__ __launch_bounds__(256) void qkv_kernel(
    const float* __restrict__ x,
    const float* __restrict__ Wq, const float* __restrict__ bq,
    const float* __restrict__ Wk, const float* __restrict__ bk,
    const float* __restrict__ Wv, const float* __restrict__ bv,
    bf16_t* __restrict__ Qb, bf16_t* __restrict__ Kb, bf16_t* __restrict__ Vt) {
  const int which = blockIdx.z;
  const float* W    = (which == 0) ? Wq : (which == 1) ? Wk : Wv;
  const float* bias = (which == 0) ? bq : (which == 1) ? bk : bv;

  __shared__ __align__(16) bf16_t As[128 * 32];
  __shared__ __align__(16) bf16_t Bs[128 * 32];

  const int tid = threadIdx.x;
  const int lane = tid & 31, wvid = tid >> 5;
  const int wm = wvid >> 2, wn = wvid & 3;          // wave tile 64m x 32n
  const int m0 = blockIdx.x * 128, n0 = blockIdx.y * 128;

  const int srow = tid >> 3, scol = (tid & 7) * 4;  // staging slot geometry

  f32x8 acc[4][2] = {};

  for (int k0 = 0; k0 < Dc; k0 += 32) {
    // stage A and B: 128x32 fp32 each -> bf16 (1024 float4 slots, 4/thread)
    for (int i = 0; i < 4; ++i) {
      int s = tid + i * 256;
      int row = s >> 3, col = (s & 7) * 4;
      float4 fa = *(const float4*)(x + (size_t)(m0 + row) * Dc + k0 + col);
      float4 fb = *(const float4*)(W + (size_t)(n0 + row) * Dc + k0 + col);
      bf16_t* da = As + row * 32 + col;
      bf16_t* db = Bs + row * 32 + col;
      da[0] = (bf16_t)fa.x; da[1] = (bf16_t)fa.y; da[2] = (bf16_t)fa.z; da[3] = (bf16_t)fa.w;
      db[0] = (bf16_t)fb.x; db[1] = (bf16_t)fb.y; db[2] = (bf16_t)fb.z; db[3] = (bf16_t)fb.w;
    }
    if (k0 + 32 < Dc) {   // hint next K-tile into cache (global_prefetch_b8)
      __builtin_prefetch(x + (size_t)(m0 + srow) * Dc + k0 + 32 + scol, 0, 1);
      __builtin_prefetch(W + (size_t)(n0 + srow) * Dc + k0 + 32 + scol, 0, 1);
    }
    __syncthreads();
    bf16x16 b0 = ld_frag(Bs, wn * 32,      32, 0, lane);
    bf16x16 b1 = ld_frag(Bs, wn * 32 + 16, 32, 0, lane);
#pragma unroll
    for (int ti = 0; ti < 4; ++ti) {
      bf16x16 a = ld_frag(As, wm * 64 + ti * 16, 32, 0, lane);
      acc[ti][0] = wmma_bf16(a, b0, acc[ti][0]);
      acc[ti][1] = wmma_bf16(a, b1, acc[ti][1]);
    }
    __syncthreads();
  }

  // epilogue: bias add, convert to bf16, scatter to head layout
  const int c = lane >> 4, nl = lane & 15;
  for (int ti = 0; ti < 4; ++ti)
    for (int tj = 0; tj < 2; ++tj)
      for (int r = 0; r < 8; ++r) {
        int m = m0 + wm * 64 + ti * 16 + r + 8 * c;   // token index (b*L + l)
        int n = n0 + wn * 32 + tj * 16 + nl;          // feature index (h*P + p)
        float v = acc[ti][tj][r] + bias[n];
        int b = m >> 11, l = m & (Lc - 1);
        int h = n >> 6,  p = n & (Pc - 1);
        bf16_t bv16 = (bf16_t)v;
        if (which == 0)
          Qb[(((size_t)b * Hc + h) * Lc + l) * Pc + p] = bv16;
        else if (which == 1)
          Kb[(((size_t)b * Hc + h) * Lc + l) * Pc + p] = bv16;
        else
          Vt[(((size_t)b * Hc + h) * Pc + p) * Lc + l] = bv16;
      }
}

// ---------------------------------------------------------------------------
// Kernel 2: streaming (flash-style) attention.
// grid = (L/128, B*H); block 256 = 8 waves; each wave owns 16 queries.
// K/V tiles staged via async global->LDS (bf16 pure copies, ASYNCcnt).
// Streams 32-key blocks: S = K.Q^T (4 WMMA), online softmax over keys,
// O += P^T.V (4 WMMA). Score D-layout feeds P^T A-frags lane-locally.
// ---------------------------------------------------------------------------
__global__ __launch_bounds__(256) void attn_kernel(
    const bf16_t* __restrict__ Qb, const bf16_t* __restrict__ Kb,
    const bf16_t* __restrict__ Vt, bf16_t* __restrict__ Ob) {
  const int bh = blockIdx.y;
  const int b = bh >> 4, h = bh & 15;
  const int tid = threadIdx.x, lane = tid & 31, wvid = tid >> 5;
  const int q0 = blockIdx.x * 128 + wvid * 16;      // wave's 16 queries
  const int m_ = lane & 15, c_ = lane >> 4;

  const bf16_t* Qp = Qb + (size_t)bh * Lc * Pc;
  const bf16_t* Kp = Kb + (size_t)bh * Lc * Pc;
  const bf16_t* Vp = Vt + (size_t)bh * Pc * Lc;

  __shared__ __align__(16) bf16_t Ks[32 * 64];   // 32 keys x 64 p
  __shared__ __align__(16) bf16_t Vs[64 * 32];   // 64 p x 32 keys (V^T)

  const unsigned ksb = (unsigned)(size_t)(void*)Ks;
  const unsigned vsb = (unsigned)(size_t)(void*)Vs;
  const int vrow = tid >> 2, vq = tid & 3;       // V^T staging geometry

  // Q fragments (B operand), register resident for the whole stream
  bf16x16 qf0 = ld_frag(Qp, q0, Pc, 0, lane);
  bf16x16 qf1 = ld_frag(Qp, q0, Pc, 32, lane);

  f32x8 o[4] = {};                                  // 16q x 64p accumulator
  float mrun = -1e30f, lrun = 0.f;

  for (int l0 = 0; l0 < Lc; l0 += 32) {
    // async stage: K block (contiguous) and V^T block (strided rows)
    async_g2lds_b128(ksb + tid * 16, Kp + (size_t)l0 * Pc + tid * 8);
    async_g2lds_b128(vsb + vrow * 64 + vq * 16,
                     Vp + (size_t)vrow * Lc + l0 + vq * 8);
    wait_async0();
    __syncthreads();

    // S[l, m] = sum_p K[l,p] Q[m,p], contraction p = 64 in two chunks
    f32x8 s0 = {}, s1 = {};
    {
      bf16x16 ka = ld_frag(Ks, 0, 64, 0, lane);
      bf16x16 kb = ld_frag(Ks, 16, 64, 0, lane);
      s0 = wmma_bf16(ka, qf0, s0);
      s1 = wmma_bf16(kb, qf0, s1);
      ka = ld_frag(Ks, 0, 64, 32, lane);
      kb = ld_frag(Ks, 16, 64, 32, lane);
      s0 = wmma_bf16(ka, qf1, s0);
      s1 = wmma_bf16(kb, qf1, s1);
    }

    // online softmax: lane holds query column m_, keys {8c_+r} and {16+8c_+r}
    const float scale = 0.125f;                     // 1/sqrt(64)
    float lmax = -1e30f;
    for (int r = 0; r < 8; ++r) {
      s0[r] *= scale; s1[r] *= scale;
      lmax = fmaxf(lmax, fmaxf(s0[r], s1[r]));
    }
    lmax = fmaxf(lmax, __shfl_xor(lmax, 16, 32));
    float mnew  = fmaxf(mrun, lmax);
    float alpha = __expf(mrun - mnew);
    float psum = 0.f;
    union { bf16x16 v; unsigned u[8]; } af;         // P^T A-frag, lane-local pack
    for (int j = 0; j < 4; ++j) {
      float e0 = __expf(s0[2 * j] - mnew), e1 = __expf(s0[2 * j + 1] - mnew);
      psum += e0 + e1; af.u[j] = pack2bf(e0, e1);
    }
    for (int j = 0; j < 4; ++j) {
      float e0 = __expf(s1[2 * j] - mnew), e1 = __expf(s1[2 * j + 1] - mnew);
      psum += e0 + e1; af.u[4 + j] = pack2bf(e0, e1);
    }
    psum += __shfl_xor(psum, 16, 32);
    lrun = lrun * alpha + psum;
    mrun = mnew;

    // rescale O: O D-layout vgpr r belongs to query (8*c_ + r)
    float av[8];
    for (int r = 0; r < 8; ++r) av[r] = __shfl(alpha, 8 * c_ + r, 32);
    for (int t = 0; t < 4; ++t)
      for (int r = 0; r < 8; ++r) o[t][r] *= av[r];

    // O += P^T . V   (4 p-tiles, K = 32 keys)
    for (int t = 0; t < 4; ++t) {
      bf16x16 vf = ld_frag(Vs, t * 16, 32, 0, lane);
      o[t] = wmma_bf16(af.v, vf, o[t]);
    }
    __syncthreads();
  }

  // normalize by running sum and store as [B, L, D] bf16 (d = h*64 + p)
  float inv = 1.f / lrun;
  float iv[8];
  for (int r = 0; r < 8; ++r) iv[r] = __shfl(inv, 8 * c_ + r, 32);
  for (int t = 0; t < 4; ++t)
    for (int r = 0; r < 8; ++r) {
      int mq = q0 + 8 * c_ + r;
      int d  = h * Pc + t * 16 + m_;
      Ob[((size_t)b * Lc + mq) * Dc + d] = (bf16_t)(o[t][r] * iv[r]);
    }
}

// ---------------------------------------------------------------------------
// Kernel 3: output projection.  out = Ob(bf16) @ Wo^T + bo  -> fp32
// grid = (M/128, N/128); WG tile 128x128, wave tile 64x32.
// A tile staged via async global->LDS (bf16 pure copy).
// ---------------------------------------------------------------------------
__global__ __launch_bounds__(256) void oproj_kernel(
    const bf16_t* __restrict__ A, const float* __restrict__ Wo,
    const float* __restrict__ bo, float* __restrict__ out) {
  __shared__ __align__(16) bf16_t As[128 * 32];
  __shared__ __align__(16) bf16_t Bs[128 * 32];

  const int tid = threadIdx.x;
  const int lane = tid & 31, wvid = tid >> 5;
  const int wm = wvid >> 2, wn = wvid & 3;
  const int m0 = blockIdx.x * 128, n0 = blockIdx.y * 128;

  const unsigned asb = (unsigned)(size_t)(void*)As;
  const int srow = tid >> 3, scol = (tid & 7) * 4;

  f32x8 acc[4][2] = {};

  for (int k0 = 0; k0 < Dc; k0 += 32) {
    // stage A: 128x32 bf16 via async copies (512 uint4 slots, 2 per thread)
    for (int i = 0; i < 2; ++i) {
      int s = tid + i * 256;
      int row = s >> 2, col = (s & 3) * 8;
      async_g2lds_b128(asb + row * 64 + (s & 3) * 16,
                       A + (size_t)(m0 + row) * Dc + k0 + col);
    }
    // stage B: 128x32 fp32 -> bf16 (1024 float4 slots, 4 per thread)
    for (int i = 0; i < 4; ++i) {
      int s = tid + i * 256;
      int row = s >> 3, col = (s & 7) * 4;
      float4 f = *(const float4*)(Wo + (size_t)(n0 + row) * Dc + k0 + col);
      bf16_t* d = Bs + row * 32 + col;
      d[0] = (bf16_t)f.x; d[1] = (bf16_t)f.y; d[2] = (bf16_t)f.z; d[3] = (bf16_t)f.w;
    }
    if (k0 + 32 < Dc) {
      __builtin_prefetch(A  + (size_t)(m0 + srow) * Dc + k0 + 32 + scol, 0, 1);
      __builtin_prefetch(Wo + (size_t)(n0 + srow) * Dc + k0 + 32 + scol, 0, 1);
    }
    wait_async0();
    __syncthreads();
    bf16x16 b0 = ld_frag(Bs, wn * 32,      32, 0, lane);
    bf16x16 b1 = ld_frag(Bs, wn * 32 + 16, 32, 0, lane);
#pragma unroll
    for (int ti = 0; ti < 4; ++ti) {
      bf16x16 a = ld_frag(As, wm * 64 + ti * 16, 32, 0, lane);
      acc[ti][0] = wmma_bf16(a, b0, acc[ti][0]);
      acc[ti][1] = wmma_bf16(a, b1, acc[ti][1]);
    }
    __syncthreads();
  }

  const int c = lane >> 4, nl = lane & 15;
  for (int ti = 0; ti < 4; ++ti)
    for (int tj = 0; tj < 2; ++tj)
      for (int r = 0; r < 8; ++r) {
        int m = m0 + wm * 64 + ti * 16 + r + 8 * c;
        int n = n0 + wn * 32 + tj * 16 + nl;
        out[(size_t)m * Dc + n] = acc[ti][tj][r] + bo[n];
      }
}

// ---------------------------------------------------------------------------
extern "C" void kernel_launch(void* const* d_in, const int* in_sizes, int n_in,
                              void* d_out, int out_size, void* d_ws, size_t ws_size,
                              hipStream_t stream) {
  const float* x  = (const float*)d_in[0];
  const float* Wq = (const float*)d_in[1];
  const float* bq = (const float*)d_in[2];
  const float* Wk = (const float*)d_in[3];
  const float* bk = (const float*)d_in[4];
  const float* Wv = (const float*)d_in[5];
  const float* bv = (const float*)d_in[6];
  const float* Wo = (const float*)d_in[7];
  const float* bo = (const float*)d_in[8];

  char* ws = (char*)d_ws;
  bf16_t* Qb = (bf16_t*)(ws);
  bf16_t* Kb = (bf16_t*)(ws + QKV_BYTES);
  bf16_t* Vt = (bf16_t*)(ws + 2 * QKV_BYTES);
  bf16_t* Ob = (bf16_t*)(ws + 3 * QKV_BYTES);

  const int M = Bc * Lc;                       // 4096
  qkv_kernel<<<dim3(M / 128, Dc / 128, 3), 256, 0, stream>>>(
      x, Wq, bq, Wk, bk, Wv, bv, Qb, Kb, Vt);
  attn_kernel<<<dim3(Lc / 128, Bc * Hc), 256, 0, stream>>>(Qb, Kb, Vt, Ob);
  oproj_kernel<<<dim3(M / 128, Dc / 128), 256, 0, stream>>>(Ob, Wo, bo, (float*)d_out);
}
